// TalkingHeadAttn_64330020159872
// MI455X (gfx1250) — matrix operand order repaired and checked by reading, not compiled, not run on hardware
//
#include <hip/hip_runtime.h>
#include <stdint.h>

typedef _Float16 f16;
typedef __attribute__((ext_vector_type(16))) _Float16 v16h;
typedef __attribute__((ext_vector_type(4)))  _Float16 v4h;
typedef __attribute__((ext_vector_type(8)))  float    v8f;
typedef __attribute__((ext_vector_type(4))) unsigned int v4u;
typedef __attribute__((ext_vector_type(8))) int          v8i;
typedef __attribute__((ext_vector_type(4))) int          v4i;

constexpr int kB = 8;
constexpr int kN = 1024;
constexpr int kC = 768;
constexpr int kH = 12;
constexpr int kD = 64;
constexpr float kScale = 0.125f;  // 64^-0.5

__device__ inline v8f wmma16x16x32(v16h a, v16h b, v8f c) {
  return __builtin_amdgcn_wmma_f32_16x16x32_f16(false, a, false, b, (short)0, c,
                                                false, false);
}

__device__ inline v8f zero8() {
  v8f z;
#pragma unroll
  for (int i = 0; i < 8; ++i) z[i] = 0.0f;
  return z;
}

// A-fragment K index for 16-bit A (16x32), ISA 7.12.2.
__device__ inline int a_kidx(int e, int half) {
  return e + 8 * half + (e >= 8 ? 8 : 0);
}

// Pack two fp32 into one dword of two f16 for 4B LDS stores.
union PackHH { f16 h[2]; unsigned u; };
__device__ inline unsigned pack2h(float a, float b) {
  PackHH p;
  p.h[0] = (f16)a;
  p.h[1] = (f16)b;
  return p.u;
}

// 12-head mix: 24 contiguous bytes of f16 scores -> fp32 dot with weight row.
__device__ inline float mix12(const f16* sp, const float* wrow, float bias) {
  const v4h* p = (const v4h*)sp;
  v4h a0 = p[0], a1 = p[1], a2 = p[2];
  float a = bias;
#pragma unroll
  for (int i = 0; i < 4; ++i) a += (float)a0[i] * wrow[i];
#pragma unroll
  for (int i = 0; i < 4; ++i) a += (float)a1[i] * wrow[4 + i];
#pragma unroll
  for (int i = 0; i < 4; ++i) a += (float)a2[i] * wrow[8 + i];
  return a;
}

// ---------------------------------------------------------------------------
// Tensor Data Mover: 2D f16 tile load, global -> LDS (D# per ISA ch.8).
// ---------------------------------------------------------------------------
#if __has_builtin(__builtin_amdgcn_tensor_load_to_lds)
#define HAS_TDM 1
#else
#define HAS_TDM 0
#endif

#if HAS_TDM
__device__ inline void tdm_load_2d_f16(unsigned lds_addr, unsigned ga_lo,
                                       unsigned ga_hi, unsigned tensor_d0,
                                       unsigned tensor_d1, unsigned tile_d0,
                                       unsigned tile_d1, unsigned stride0) {
  v4u g0;
  g0[0] = 1u;                                       // count=1 (valid, user)
  g0[1] = lds_addr;                                 // lds_addr (bytes)
  g0[2] = ga_lo;                                    // global_addr[31:0]
  g0[3] = (ga_hi & 0x1FFFFFFu) | (2u << 30);        // addr[56:32] | type=2
  v8i g1;
  g1[0] = (int)(1u << 16);                          // data_size = 2 bytes
  g1[1] = (int)((tensor_d0 & 0xFFFFu) << 16);       // tensor_dim0[15:0]
  g1[2] = (int)(((tensor_d0 >> 16) & 0xFFFFu) |
                ((tensor_d1 & 0xFFFFu) << 16));     // dim0 hi | dim1 lo
  g1[3] = (int)(((tensor_d1 >> 16) & 0xFFFFu) |
                ((tile_d0 & 0xFFFFu) << 16));       // dim1 hi | tile_dim0
  g1[4] = (int)(tile_d1 & 0xFFFFu);                 // tile_dim1 (tile_dim2=0)
  g1[5] = (int)stride0;                             // tensor_dim0_stride lo
  g1[6] = 0;
  g1[7] = 0;
  v4i z4 = {0, 0, 0, 0};
#if defined(__clang_major__) && (__clang_major__ >= 23)
  v8i z8 = {0, 0, 0, 0, 0, 0, 0, 0};
  __builtin_amdgcn_tensor_load_to_lds(g0, g1, z4, z4, z8, 0);
#else
  __builtin_amdgcn_tensor_load_to_lds(g0, g1, z4, z4, 0);
#endif
}
#endif

__device__ inline void wait_tensorcnt0() {
#if __has_builtin(__builtin_amdgcn_s_wait_tensorcnt)
  __builtin_amdgcn_s_wait_tensorcnt(0);
#elif HAS_TDM
  asm volatile("s_wait_tensorcnt 0" ::: "memory");
#endif
}

// ---------------------------------------------------------------------------
// Kernel 1: QKV projection. x(8192x768) @ w_qkv(768x2304) + b ->
// Q (scaled), K, V stored f16 as (B,H,N,D).
// ---------------------------------------------------------------------------
__global__ __launch_bounds__(256) void qkv_gemm_kernel(
    const float* __restrict__ x, const float* __restrict__ wqkv,
    const float* __restrict__ bqkv, f16* __restrict__ qg, f16* __restrict__ kg,
    f16* __restrict__ vg) {
  __shared__ __align__(32) f16 As[128][40];      // 128x32 tile (row-major)
  __shared__ __align__(32) f16 Bsf[4][32][16];   // frag-order: [j][lane][e]

  const int t = threadIdx.x;
  const int wave = t >> 5;
  const int lane = t & 31;
  const int half = lane >> 4;
  const int l15 = lane & 15;
  const int m0 = blockIdx.x * 128;
  const int n0 = blockIdx.y * 64;

  v8f acc[4];
#pragma unroll
  for (int j = 0; j < 4; ++j) acc[j] = zero8();

  for (int kk = 0; kk < kC; kk += 32) {
    // A tile: pairs of fp32 -> one packed dword LDS store (2048 pairs).
#pragma unroll
    for (int i = 0; i < 8; ++i) {
      int p = i * 256 + t;
      int r = p >> 4, cp = (p & 15) * 2;
      const float* src = &x[(size_t)(m0 + r) * kC + kk + cp];
      *(unsigned*)&As[r][cp] = pack2h(src[0], src[1]);
    }
    // B tile: global (r=K row 0..31, c=N col 0..63) -> fragment order.
#pragma unroll
    for (int i = 0; i < 8; ++i) {
      int idx = i * 256 + t;
      int r = idx >> 6, c = idx & 63;
      f16 v = (f16)wqkv[(size_t)(kk + r) * (3 * kC) + n0 + c];
      Bsf[c >> 4][(c & 15) + 16 * (r >> 4)][r & 15] = v;
    }
    __syncthreads();

    v16h a;
#pragma unroll
    for (int e = 0; e < 16; ++e) a[e] = As[wave * 16 + l15][e + 16 * half];
#pragma unroll
    for (int j = 0; j < 4; ++j) {
      v16h bf = *(const v16h*)&Bsf[j][lane][0];
      acc[j] = wmma16x16x32(a, bf, acc[j]);
    }
    __syncthreads();
  }

  // Epilogue. The 64-column span of this block sits inside exactly one of
  // Q/K/V and one head: which/h are block-uniform scalars.
  const int which = n0 / kC;
  const int hh = (n0 % kC) >> 6;
  f16* dst = (which == 0) ? qg : ((which == 1) ? kg : vg);
  const float mul = (which == 0) ? kScale : 1.0f;
#pragma unroll
  for (int j = 0; j < 4; ++j) {
    const float bias = bqkv[n0 + j * 16 + l15];
#pragma unroll
    for (int i = 0; i < 8; ++i) {
      int r = m0 + wave * 16 + i + 8 * half;
      int d = j * 16 + l15;
      int bidx = r >> 10, nidx = r & 1023;
      float v = (acc[j][i] + bias) * mul;
      dst[(((size_t)bidx * kH + hh) * kN + nidx) * kD + d] = (f16)v;
    }
  }
}

// ---------------------------------------------------------------------------
// Kernel 2: fused talking-heads attention. Block = (batch, 16 queries),
// 12 waves = one head each. Two passes over 32-key tiles (flash-style).
// Score exchange buffer is head-innermost so each mix position is 3 x b64
// LDS loads. V tiles are DMA'd into LDS by the Tensor Data Mover.
// ---------------------------------------------------------------------------
__global__ __launch_bounds__(384) void attn_kernel(
    const f16* __restrict__ qg, const f16* __restrict__ kg,
    const f16* __restrict__ vg, const float* __restrict__ wl,
    const float* __restrict__ bl, const float* __restrict__ ww,
    const float* __restrict__ bw, float* __restrict__ ao) {
  __shared__ __align__(32) f16 Sl2[16][32][kH];   // [row][col][head] S/P
  __shared__ __align__(32) f16 Vbuf[kH][32][64];  // per-head TDM V tile (4KB)
  __shared__ float wlsT[kH * kH], wwsT[kH * kH], bls[kH], bws[kH];

  const int t = threadIdx.x;
  const int g = __builtin_amdgcn_readfirstlane(t >> 5);  // wave's head
  const int lane = t & 31;
  const int half = lane >> 4;
  const int l15 = lane & 15;
  const int b = blockIdx.y;
  const int q0 = blockIdx.x * 16;

  if (t < kH * kH) {
    // Transposed so each wave's weight column is contiguous.
    wlsT[(t % kH) * kH + (t / kH)] = wl[t];
    wwsT[(t % kH) * kH + (t / kH)] = ww[t];
  }
  if (t < kH) { bls[t] = bl[t]; bws[t] = bw[t]; }
  __syncthreads();

  const size_t headBase = ((size_t)b * kH + g) * kN;
  const float* wlrow = &wlsT[g * kH];
  const float* wwrow = &wwsT[g * kH];

  // Q fragments (16 queries x 64 dims -> two K=32 A-frags), kernel-resident.
  v16h qa[2];
#pragma unroll
  for (int f = 0; f < 2; ++f) {
#pragma unroll
    for (int e = 0; e < 16; ++e) {
      int dim = f * 32 + e + 16 * half;
      qa[f][e] = qg[(headBase + q0 + l15) * kD + dim];
    }
  }

  float mrun[8], lrun[8];
#pragma unroll
  for (int i = 0; i < 8; ++i) { mrun[i] = -1e30f; lrun[i] = 0.0f; }

  // ---------------- Pass A: softmax statistics ----------------
  for (int k0 = 0; k0 < kN; k0 += 32) {
    if (k0 + 32 < kN)
      __builtin_prefetch(&kg[(headBase + k0 + 32 + l15) * kD], 0, 1);

    v8f s[2];
    s[0] = zero8(); s[1] = zero8();
#pragma unroll
    for (int kb = 0; kb < 2; ++kb) {
#pragma unroll
      for (int f = 0; f < 2; ++f) {
        v16h kf = *(const v16h*)&kg[(headBase + k0 + kb * 16 + l15) * kD +
                                    f * 32 + 16 * half];
        s[kb] = wmma16x16x32(qa[f], kf, s[kb]);
      }
    }
#pragma unroll
    for (int kb = 0; kb < 2; ++kb)
#pragma unroll
      for (int i = 0; i < 8; ++i)
        Sl2[i + 8 * half][kb * 16 + l15][g] = (f16)s[kb][i];
    __syncthreads();

    // Pre-softmax talking-heads mix at this wave's C-layout positions.
    float av[2][8];
#pragma unroll
    for (int kb = 0; kb < 2; ++kb)
#pragma unroll
      for (int i = 0; i < 8; ++i)
        av[kb][i] =
            mix12(&Sl2[i + 8 * half][kb * 16 + l15][0], wlrow, bls[g]);
#pragma unroll
    for (int i = 0; i < 8; ++i) {
      float tmax = fmaxf(av[0][i], av[1][i]);
#pragma unroll
      for (int off = 8; off >= 1; off >>= 1)
        tmax = fmaxf(tmax, __shfl_xor(tmax, off, 16));
      float mnew = fmaxf(mrun[i], tmax);
      float ps = __expf(av[0][i] - mnew) + __expf(av[1][i] - mnew);
#pragma unroll
      for (int off = 8; off >= 1; off >>= 1) ps += __shfl_xor(ps, off, 16);
      lrun[i] = lrun[i] * __expf(mrun[i] - mnew) + ps;
      mrun[i] = mnew;
    }
    __syncthreads();
  }

  // ---------------- Pass B: normalized P, post-mix, PV ----------------
  v8f oacc[4];
#pragma unroll
  for (int f = 0; f < 4; ++f) oacc[f] = zero8();

  for (int k0 = 0; k0 < kN; k0 += 32) {
    // Kick off the V-tile DMA for this step (32 keys x 64 dims, contiguous).
    {
      unsigned long long ga =
          (unsigned long long)(uintptr_t)(vg + (headBase + k0) * kD);
#if HAS_TDM
      unsigned ldsA = (unsigned)__builtin_amdgcn_readfirstlane(
          (int)(uintptr_t)&Vbuf[g][0][0]);
      unsigned galo =
          (unsigned)__builtin_amdgcn_readfirstlane((int)(ga & 0xFFFFFFFFull));
      unsigned gahi = (unsigned)__builtin_amdgcn_readfirstlane((int)(ga >> 32));
      tdm_load_2d_f16(ldsA, galo, gahi, /*tensor_d0=*/kD, /*tensor_d1=*/kN,
                      /*tile_d0=*/kD, /*tile_d1=*/32, /*stride0=*/kD);
#else
      f16* dst = &Vbuf[g][0][0];
      const f16* src = vg + (headBase + k0) * kD;
      for (int i = lane; i < 32 * kD; i += 32) dst[i] = src[i];
#endif
    }

    // Recompute S_h for this tile (overlaps with the TDM transfer).
    v8f s[2];
    s[0] = zero8(); s[1] = zero8();
#pragma unroll
    for (int kb = 0; kb < 2; ++kb) {
#pragma unroll
      for (int f = 0; f < 2; ++f) {
        v16h kf = *(const v16h*)&kg[(headBase + k0 + kb * 16 + l15) * kD +
                                    f * 32 + 16 * half];
        s[kb] = wmma16x16x32(qa[f], kf, s[kb]);
      }
    }
#pragma unroll
    for (int kb = 0; kb < 2; ++kb)
#pragma unroll
      for (int i = 0; i < 8; ++i)
        Sl2[i + 8 * half][kb * 16 + l15][g] = (f16)s[kb][i];
    __syncthreads();

    // Pre-mix + normalized softmax value P_g.
    float pv[2][8];
#pragma unroll
    for (int kb = 0; kb < 2; ++kb)
#pragma unroll
      for (int i = 0; i < 8; ++i) {
        float a = mix12(&Sl2[i + 8 * half][kb * 16 + l15][0], wlrow, bls[g]);
        pv[kb][i] = __expf(a - mrun[i]) / lrun[i];
      }
    __syncthreads();  // all heads done reading S before overwriting with P
#pragma unroll
    for (int kb = 0; kb < 2; ++kb)
#pragma unroll
      for (int i = 0; i < 8; ++i)
        Sl2[i + 8 * half][kb * 16 + l15][g] = (f16)pv[kb][i];
    __syncthreads();

    // Post-softmax mix gathered directly at A-fragment positions (K=32 keys).
    v16h pa;
#pragma unroll
    for (int e = 0; e < 16; ++e) {
      int key = e + 16 * half;
      pa[e] = (f16)mix12(&Sl2[l15][key][0], wwrow, bws[g]);
    }

    // V tile is in LDS by now (TDM); build B-frags and accumulate.
    wait_tensorcnt0();
#pragma unroll
    for (int f = 0; f < 4; ++f) {
      v16h vb;
#pragma unroll
      for (int e = 0; e < 16; ++e)
        vb[e] = Vbuf[g][e + 16 * half][f * 16 + l15];
      oacc[f] = wmma16x16x32(pa, vb, oacc[f]);
    }
    __syncthreads();
  }

  // Write (B,N,H*D) fp32.
#pragma unroll
  for (int f = 0; f < 4; ++f) {
#pragma unroll
    for (int i = 0; i < 8; ++i) {
      int row = q0 + i + 8 * half;
      int dd = f * 16 + l15;
      ao[((size_t)b * kN + row) * kC + g * kD + dd] = oacc[f][i];
    }
  }
}

// ---------------------------------------------------------------------------
// Kernel 3: output projection. ao(8192x768 fp32) @ w_proj(768x768) + b -> out.
// ---------------------------------------------------------------------------
__global__ __launch_bounds__(256) void proj_gemm_kernel(
    const float* __restrict__ ain, const float* __restrict__ wp,
    const float* __restrict__ bp, float* __restrict__ out) {
  __shared__ __align__(32) f16 As[128][40];
  __shared__ __align__(32) f16 Bsf[4][32][16];

  const int t = threadIdx.x;
  const int wave = t >> 5;
  const int lane = t & 31;
  const int half = lane >> 4;
  const int l15 = lane & 15;
  const int m0 = blockIdx.x * 128;
  const int n0 = blockIdx.y * 64;

  v8f acc[4];
#pragma unroll
  for (int j = 0; j < 4; ++j) acc[j] = zero8();

  for (int kk = 0; kk < kC; kk += 32) {
#pragma unroll
    for (int i = 0; i < 8; ++i) {
      int p = i * 256 + t;
      int r = p >> 4, cp = (p & 15) * 2;
      const float* src = &ain[(size_t)(m0 + r) * kC + kk + cp];
      *(unsigned*)&As[r][cp] = pack2h(src[0], src[1]);
    }
#pragma unroll
    for (int i = 0; i < 8; ++i) {
      int idx = i * 256 + t;
      int r = idx >> 6, c = idx & 63;
      f16 v = (f16)wp[(size_t)(kk + r) * kC + n0 + c];
      Bsf[c >> 4][(c & 15) + 16 * (r >> 4)][r & 15] = v;
    }
    __syncthreads();

    v16h a;
#pragma unroll
    for (int e = 0; e < 16; ++e) a[e] = As[wave * 16 + l15][e + 16 * half];
#pragma unroll
    for (int j = 0; j < 4; ++j) {
      v16h bf = *(const v16h*)&Bsf[j][lane][0];
      acc[j] = wmma16x16x32(a, bf, acc[j]);
    }
    __syncthreads();
  }

#pragma unroll
  for (int j = 0; j < 4; ++j) {
    const float bias = bp[n0 + j * 16 + l15];
#pragma unroll
    for (int i = 0; i < 8; ++i) {
      int r = m0 + wave * 16 + i + 8 * half;
      int c = n0 + j * 16 + l15;
      out[(size_t)r * kC + c] = acc[j][i] + bias;
    }
  }
}

// ---------------------------------------------------------------------------
extern "C" void kernel_launch(void* const* d_in, const int* in_sizes, int n_in,
                              void* d_out, int out_size, void* d_ws,
                              size_t ws_size, hipStream_t stream) {
  const float* x     = (const float*)d_in[0];
  const float* wqkv  = (const float*)d_in[1];
  const float* bqkv  = (const float*)d_in[2];
  const float* wl    = (const float*)d_in[3];
  const float* bl    = (const float*)d_in[4];
  const float* ww    = (const float*)d_in[5];
  const float* bw    = (const float*)d_in[6];
  const float* wproj = (const float*)d_in[7];
  const float* bproj = (const float*)d_in[8];
  float* out = (float*)d_out;

  const size_t qkvElems = (size_t)kB * kH * kN * kD;  // 6,291,456
  f16* qg = (f16*)d_ws;
  f16* kg = qg + qkvElems;
  f16* vg = kg + qkvElems;
  float* ao = (float*)(vg + qkvElems);

  qkv_gemm_kernel<<<dim3(8192 / 128, (3 * kC) / 64), 256, 0, stream>>>(
      x, wqkv, bqkv, qg, kg, vg);

  attn_kernel<<<dim3(kN / 16, kB), 12 * 32, 0, stream>>>(qg, kg, vg, wl, bl,
                                                         ww, bw, ao);

  proj_gemm_kernel<<<dim3(8192 / 128, kC / 64), 256, 0, stream>>>(ao, wproj,
                                                                  bproj, out);
}
